// GraphCNN_43980465111235
// MI455X (gfx1250) — matrix operand order repaired and checked by reading, not compiled, not run on hardware
//
#include <hip/hip_runtime.h>
#include <hip/hip_bf16.h>

typedef _Float16 half_t;
typedef __attribute__((ext_vector_type(16))) _Float16 v16h;
typedef __attribute__((ext_vector_type(8)))  _Float16 v8h;
typedef __attribute__((ext_vector_type(8)))  float    v8f;

// ---------------------------------------------------------------------------
// Generic f16 WMMA GEMM:  C[M,N](f16) = A[M,K](f16) x B[K,N](f16) + bias + res
//   block tile 128(M) x 64(N), K staged 64 deep in LDS, 8 wave32 waves,
//   each wave = 32x32 (2x2 tiles of v_wmma_f32_16x16x32_f16, 2 K sub-steps
//   per stage => 8 WMMA per barrier pair).
//   A staged row-major (b128 loads), B staged transposed via in-register
//   transpose + LDS b16 scatter (coalesced b128 global loads).
//   bias index = row / bias_div (fp32). res: optional f16, same ldc.
//   stores skipped when (col % col_mod) >= col_valid  (vertex padding).
//   Requires: K % 64 == 0, N % 64 == 0.
// ---------------------------------------------------------------------------
__global__ __launch_bounds__(256) void gemm_wmma_f16(
    const half_t* __restrict__ A, const half_t* __restrict__ Bm,
    half_t* __restrict__ C,
    const float* __restrict__ bias, int bias_div,
    const half_t* __restrict__ res,
    int M, int N, int K, int lda, int ldb, int ldc,
    int col_mod, int col_valid)
{
    __shared__ __align__(32) half_t As[128 * 64];   // [row][k]   16 KB
    __shared__ __align__(32) half_t Bs[64 * 64];    // [col][k]    8 KB

    const int tid  = threadIdx.x;
    const int lane = tid & 31;
    const int wave = tid >> 5;
    const int wm   = wave & 3;   // 4 M-subtiles of 32
    const int wn   = wave >> 2;  // 2 N-subtiles of 32
    const int m0   = blockIdx.y * 128;
    const int n0   = blockIdx.x * 64;

    v8f acc[2][2] = {};

    const int kbA = (lane < 16) ? 0 : 16;  // A frag K base within 32-chunk
    const int kbB = (lane < 16) ? 0 : 16;  // B frag K base within 32-chunk

    for (int k0 = 0; k0 < K; k0 += 64) {
        // ---- stage A tile (128 rows x 64 k), vector b128 copies ----
#pragma unroll
        for (int i = 0; i < 4; ++i) {
            int c   = tid + i * 256;        // 1024 chunks of 8 halfs
            int row = c >> 3;
            int seg = (c & 7) * 8;
            v8h v = {};
            int gr = m0 + row;
            if (gr < M)
                v = *(const v8h*)(A + (size_t)gr * lda + k0 + seg);
            *(v8h*)(As + row * 64 + seg) = v;
        }
        // ---- stage B tile (64 k x 64 cols) -> transposed Bs[col][k] ----
        //      coalesced b128 global loads + in-register transpose scatter
#pragma unroll
        for (int i = 0; i < 2; ++i) {
            int c    = tid + i * 256;       // 512 chunks of 8 halfs
            int krow = c >> 3;
            int seg  = (c & 7) * 8;
            v8h v = *(const v8h*)(Bm + (size_t)(k0 + krow) * ldb + n0 + seg);
#pragma unroll
            for (int e = 0; e < 8; ++e)
                Bs[(seg + e) * 64 + krow] = v[e];
        }
        __syncthreads();

        // ---- 2 K sub-steps of 32: fragments in contiguous k order (k = 16h + s, A and B alike) ----
#pragma unroll
        for (int s = 0; s < 2; ++s) {
            const int ko = s * 32;
            v16h af[2], bf[2];
#pragma unroll
            for (int mt = 0; mt < 2; ++mt) {
                int r = wm * 32 + mt * 16 + (lane & 15);
                v8h lo = *(const v8h*)(As + r * 64 + ko + kbA);      // K kb..kb+7
                v8h hi = *(const v8h*)(As + r * 64 + ko + kbA + 8);  // K kb+8..kb+15
                af[mt] = __builtin_shufflevector(lo, hi,
                           0,1,2,3,4,5,6,7, 8,9,10,11,12,13,14,15);
            }
#pragma unroll
            for (int nt = 0; nt < 2; ++nt) {
                int cl = wn * 32 + nt * 16 + (lane & 15);
                bf[nt] = *(const v16h*)(Bs + cl * 64 + ko + kbB);    // K kbB..kbB+15
            }
#pragma unroll
            for (int mt = 0; mt < 2; ++mt)
#pragma unroll
                for (int nt = 0; nt < 2; ++nt)
                    acc[mt][nt] = __builtin_amdgcn_wmma_f32_16x16x32_f16(
                        false, af[mt], false, bf[nt],
                        (short)0, acc[mt][nt], false, false);
        }
        __syncthreads();
    }

    // ---- epilogue: C/D layout: VGPR j -> M = j (lanes<16) or j+8 ----
#pragma unroll
    for (int mt = 0; mt < 2; ++mt) {
#pragma unroll
        for (int nt = 0; nt < 2; ++nt) {
            int col   = n0 + wn * 32 + nt * 16 + (lane & 15);
            bool cok  = (col % col_mod) < col_valid;
            int rbase = m0 + wm * 32 + mt * 16 + ((lane < 16) ? 0 : 8);
#pragma unroll
            for (int j = 0; j < 8; ++j) {
                int row = rbase + j;
                if (cok && row < M) {
                    float v = acc[mt][nt][j];
                    if (bias) v += bias[row / bias_div];
                    if (res)  v += (float)res[(size_t)row * ldc + col];
                    C[(size_t)row * ldc + col] = (half_t)v;
                }
            }
        }
    }
}

// ---------------------------------------------------------------------------
// GroupNorm(8 channels/group) + ReLU over f16 activations [C][B][Vp]
// one block per (group, batch)
// ---------------------------------------------------------------------------
__global__ __launch_bounds__(256) void gn_relu_f16(
    const half_t* __restrict__ x, half_t* __restrict__ y,
    const float* __restrict__ gamma, const float* __restrict__ beta,
    int BVr, int Vp, int Vv)
{
    const int b   = blockIdx.x & 7;
    const int grp = blockIdx.x >> 3;
    const int c0  = grp * 8;
    const int tid = threadIdx.x;

    float s = 0.f, q = 0.f;
    for (int c = 0; c < 8; ++c) {
        const half_t* row = x + (size_t)(c0 + c) * BVr + b * Vp;
        for (int v = tid; v < Vv; v += 256) {
            float t = (float)row[v];
            s += t; q += t * t;
        }
    }
    __shared__ float rs[256], rq[256];
    rs[tid] = s; rq[tid] = q;
    __syncthreads();
    for (int off = 128; off > 0; off >>= 1) {
        if (tid < off) { rs[tid] += rs[tid + off]; rq[tid] += rq[tid + off]; }
        __syncthreads();
    }
    const float cnt  = 8.0f * (float)Vv;
    const float mean = rs[0] / cnt;
    const float var  = rq[0] / cnt - mean * mean;
    const float inv  = rsqrtf(var + 1e-5f);

    for (int c = 0; c < 8; ++c) {
        const float ga = gamma[c0 + c] * inv;
        const float be = beta[c0 + c] - mean * ga;
        const half_t* row  = x + (size_t)(c0 + c) * BVr + b * Vp;
        half_t*       orow = y + (size_t)(c0 + c) * BVr + b * Vp;
        for (int v = tid; v < Vv; v += 256) {
            float t = (float)row[v] * ga + be;
            if (t < 0.f) t = 0.f;
            orow[v] = (half_t)t;
        }
    }
}

// fp32 -> f16 weight conversion with zero K/M padding, optional transpose
__global__ __launch_bounds__(256) void cvt_w_f16(
    const float* __restrict__ src, half_t* __restrict__ dst,
    int Mo, int Mv, int Kp, int Kv, int sld, int transp)
{
    int id = blockIdx.x * 256 + threadIdx.x;
    if (id >= Mo * Kp) return;
    int r = id / Kp, c = id % Kp;
    float v = 0.f;
    if (r < Mv && c < Kv)
        v = transp ? src[(size_t)c * sld + r] : src[(size_t)r * sld + c];
    dst[id] = (half_t)v;
}

// build X0 [2051(+pad)][B][Vp] from xyz + sampled/global image features
__global__ __launch_bounds__(256) void build_x0(
    const float* __restrict__ img, const float* __restrict__ joints,
    const float* __restrict__ nearK, const float* __restrict__ farK,
    half_t* __restrict__ X0, int V, int Vp, int N1, int BVr)
{
    int id = blockIdx.x * 256 + threadIdx.x;
    const int total = 2051 * 8 * V;
    if (id >= total) return;
    int v = id % V;
    int b = (id / V) & 7;
    int k = id / (V * 8);
    float val;
    if (k < 3) {
        val = (v < N1) ? nearK[((size_t)b * N1 + v) * 3 + k]
                       : farK[((size_t)b * (V - N1) + (v - N1)) * 3 + k];
    } else {
        val = (v < N1) ? joints[((size_t)b * N1 + v) * 2048 + (k - 3)]
                       : img[(size_t)b * 2048 + (k - 3)];
    }
    X0[(size_t)k * BVr + b * Vp + v] = (half_t)val;
}

// shape head: out[b,o,v] = bias[o] + sum_c sn[c,b,v] * w[o,c]   (fp32 out)
__global__ __launch_bounds__(256) void shape_head(
    const half_t* __restrict__ sn, const float* __restrict__ w,
    const float* __restrict__ bias, float* __restrict__ out,
    int BVr, int Vp, int V)
{
    int id = blockIdx.x * 256 + threadIdx.x;
    int total = 8 * 3 * V;
    if (id >= total) return;
    int v = id % V;
    int o = (id / V) % 3;
    int b = id / (3 * V);
    float acc = bias[o];
#pragma unroll
    for (int c = 0; c < 32; ++c)
        acc += (float)sn[(size_t)c * BVr + b * Vp + v] * w[o * 32 + c];
    out[id] = acc;
}

// camera 512->1 per vertex, relu
__global__ __launch_bounds__(256) void cam_c_kernel(
    const half_t* __restrict__ cn, const float* __restrict__ w,
    const float* __restrict__ b1, float* __restrict__ cb,
    int BVr, int Vp, int V)
{
    int id = blockIdx.x * 256 + threadIdx.x;
    if (id >= 8 * V) return;
    int v = id % V, b = id / V;
    float acc = b1[0];
    for (int c = 0; c < 512; ++c)
        acc += (float)cn[(size_t)c * BVr + b * Vp + v] * w[c];
    cb[id] = acc > 0.f ? acc : 0.f;
}

// camera FC: out[b,o] = bfc[o] + sum_v cb[b,v] * wfc[o,v]
__global__ __launch_bounds__(256) void cam_out_kernel(
    const float* __restrict__ cb, const float* __restrict__ wfc,
    const float* __restrict__ bfc, float* __restrict__ out, int V)
{
    int b = blockIdx.x / 3, o = blockIdx.x % 3;
    __shared__ float r[256];
    float acc = 0.f;
    for (int v = threadIdx.x; v < V; v += 256)
        acc += cb[b * V + v] * wfc[o * V + v];
    r[threadIdx.x] = acc;
    __syncthreads();
    for (int off = 128; off > 0; off >>= 1) {
        if (threadIdx.x < off) r[threadIdx.x] += r[threadIdx.x + off];
        __syncthreads();
    }
    if (threadIdx.x == 0) out[b * 3 + o] = r[0] + bfc[o];
}

// ---------------------------------------------------------------------------
extern "C" void kernel_launch(void* const* d_in, const int* in_sizes, int n_in,
                              void* d_out, int out_size, void* d_ws, size_t ws_size,
                              hipStream_t stream)
{
    (void)in_sizes; (void)n_in; (void)out_size;

    const int V = 1723, Vp = 1728, N1 = 300, Bc = 8;
    const int BV = Vp * Bc;            // 13824 (mult of 64)
    const int K0 = 2051, K0p = 2112;   // gc_lin K padded to mult of 64

    const float* in_img   = (const float*)d_in[0];
    const float* in_joint = (const float*)d_in[1];
    const float* in_near  = (const float*)d_in[2];
    const float* in_far   = (const float*)d_in[3];
    const float* in_A     = (const float*)d_in[4];

    int pi = 5;
    auto P = [&]() { return (const float*)d_in[pi++]; };
    const float* gcW = P(); const float* gcB = P();

    struct BP { const float *preg,*preb,*w1,*b1,*n1g,*n1b,*wc,*bc,*n2g,*n2b,*w2,*b2,*wsk,*bsk; };
    auto readBlk = [&](bool skip) {
        BP p;
        p.preg=P(); p.preb=P(); p.w1=P(); p.b1=P(); p.n1g=P(); p.n1b=P();
        p.wc=P();  p.bc=P();  p.n2g=P(); p.n2b=P(); p.w2=P(); p.b2=P();
        if (skip) { p.wsk=P(); p.bsk=P(); } else { p.wsk=nullptr; p.bsk=nullptr; }
        return p;
    };
    BP gb[6]; gb[0] = readBlk(true);
    for (int i = 1; i < 6; ++i) gb[i] = readBlk(false);
    BP sbp[2]; sbp[0] = readBlk(true); sbp[1] = readBlk(true);
    const float* shnG = P(); const float* shnB = P();
    const float* shW  = P(); const float* shB  = P();
    const float* cnG  = P(); const float* cnB  = P();
    const float* clW  = P(); const float* clB  = P();
    const float* fcW  = P(); const float* fcB  = P();

    // ---- workspace arena (row-aligned f16 buffers: row = BV halfs) ----
    half_t* H = (half_t*)d_ws;
    const size_t ROW = (size_t)BV;
    half_t* X0   = H;                    // 2112 rows (dead after gc_lin)
    half_t* tbuf = H;                    // rows [0,1024)     reuse of X0 region
    half_t* y1   = H + 1024 * ROW;       // 256 rows
    half_t* t2   = H + 1280 * ROW;       // 256 rows
    half_t* sbuf = H + 1536 * ROW;       // 256 rows
    half_t* zbuf = H + 1792 * ROW;       // 256 rows (<=2112)
    half_t* actA = H + 2112 * ROW;       // 1024 rows
    half_t* actB = H + 3136 * ROW;       // 512 rows
    half_t* skb  = H + 3648 * ROW;       // 512 rows
    half_t* shp  = H + 4160 * ROW;       // 32 rows
    half_t* snm  = H + 4192 * ROW;       // 32 rows
    size_t off   = 4224 * ROW;           // halfs

    auto alloc_h = [&](size_t n) {
        half_t* p = H + off;
        off += (n + 255) & ~(size_t)255;
        return p;
    };
    half_t* At16 = alloc_h((size_t)Vp * Vp);      // Aᵀ in f16, zero-padded
    half_t* WgcH = alloc_h((size_t)1024 * K0p);

    struct BW { half_t *w1, *wct, *w2, *wsk; int cin, cout, h, cinp, hp; };
    auto allocBlk = [&](const BP& p, int cin, int cout) {
        BW w; w.cin = cin; w.cout = cout; w.h = cout / 2;
        w.cinp = (cin + 63) & ~63; w.hp = (w.h + 63) & ~63;   // K pads mult 64
        w.w1  = alloc_h((size_t)w.h * w.cinp);
        w.wct = alloc_h((size_t)w.h * w.hp);
        w.w2  = alloc_h((size_t)cout * w.hp);
        w.wsk = p.wsk ? alloc_h((size_t)cout * w.cinp) : nullptr;
        return w;
    };
    BW gw[6];
    gw[0] = allocBlk(gb[0], 1024, 512);
    for (int i = 1; i < 6; ++i) gw[i] = allocBlk(gb[i], 512, 512);
    BW sw[2];
    sw[0] = allocBlk(sbp[0], 512, 64);
    sw[1] = allocBlk(sbp[1], 64, 32);
    float* cbuf = (float*)(H + off);
    off += ((size_t)Bc * V * 2 + 255) & ~(size_t)255;

    size_t usedBytes = off * sizeof(half_t);
    size_t zb = (ws_size && usedBytes > ws_size) ? ws_size : usedBytes;
    hipMemsetAsync(d_ws, 0, zb, stream);   // keeps padded vertex cols zero

    // ---- weight conversions ----
    auto cvt = [&](const float* src, half_t* dst, int Mo, int Mv, int Kp, int Kv,
                   int sld, int tr) {
        int total = Mo * Kp;
        cvt_w_f16<<<(total + 255) / 256, 256, 0, stream>>>(src, dst, Mo, Mv, Kp, Kv, sld, tr);
    };
    cvt(in_A, At16, Vp, V, Vp, V, V, 1);               // At16[v][u] = A[u][v]
    cvt(gcW, WgcH, 1024, 1024, K0p, K0, K0, 0);
    auto cvtBlk = [&](const BP& p, const BW& w) {
        cvt(p.w1, w.w1, w.h, w.h, w.cinp, w.cin, w.cin, 0);
        cvt(p.wc, w.wct, w.h, w.h, w.hp, w.h, w.h, 1);  // W^T
        cvt(p.w2, w.w2, w.cout, w.cout, w.hp, w.h, w.h, 0);
        if (p.wsk) cvt(p.wsk, w.wsk, w.cout, w.cout, w.cinp, w.cin, w.cin, 0);
    };
    for (int i = 0; i < 6; ++i) cvtBlk(gb[i], gw[i]);
    cvtBlk(sbp[0], sw[0]);
    cvtBlk(sbp[1], sw[1]);

    // ---- build input tensor ----
    {
        int total = 2051 * 8 * V;
        build_x0<<<(total + 255) / 256, 256, 0, stream>>>(
            in_img, in_joint, in_near, in_far, X0, V, Vp, N1, BV);
    }

    auto gemm = [&](const half_t* A, const half_t* Bm, half_t* C,
                    const float* bias, int bdiv, const half_t* res,
                    int M, int N, int K, int lda, int ldb, int ldc) {
        dim3 g(N / 64, (M + 127) / 128);
        gemm_wmma_f16<<<g, 256, 0, stream>>>(A, Bm, C, bias, bdiv, res,
                                             M, N, K, lda, ldb, ldc, Vp, V);
    };
    auto gn = [&](const half_t* x, half_t* y, const float* g, const float* b, int C) {
        gn_relu_f16<<<(C / 8) * Bc, 256, 0, stream>>>(x, y, g, b, BV, Vp, V);
    };

    // gc_lin: (1024 x 2112) x (2112 x BV) -> actA
    gemm(WgcH, X0, actA, gcB, 1, nullptr, 1024, BV, K0p, K0p, BV, BV);

    auto runBlock = [&](const BP& p, const BW& w, const half_t* xin, half_t* xout) {
        gn(xin, tbuf, p.preg, p.preb, w.cin);
        gemm(w.w1, tbuf, y1, p.b1, 1, nullptr, w.h, BV, w.cinp, w.cinp, BV, BV);
        gn(y1, t2, p.n1g, p.n1b, w.h);
        // support = W^T x
        gemm(w.wct, t2, sbuf, nullptr, 1, nullptr, w.h, BV, w.hp, w.hp, BV, BV);
        // out = A @ support  (one GEMM across batch: rows = (d,b))
        gemm(sbuf, At16, zbuf, p.bc, Bc, nullptr, w.h * Bc, Vp, Vp, Vp, Vp, Vp);
        gn(zbuf, t2, p.n2g, p.n2b, w.h);
        const half_t* res = xin;
        if (w.wsk) {
            gemm(w.wsk, xin, skb, p.bsk, 1, nullptr, w.cout, BV, w.cinp, w.cinp, BV, BV);
            res = skb;
        }
        gemm(w.w2, t2, xout, p.b2, 1, res, w.cout, BV, w.hp, w.hp, BV, BV);
    };

    half_t* cur = actA;
    half_t* nxt = actB;
    for (int i = 0; i < 6; ++i) {
        runBlock(gb[i], gw[i], cur, nxt);
        half_t* t = cur; cur = nxt; nxt = t;
    }
    // after 6 blocks: x512 lives in actA (cur)

    // shape path
    runBlock(sbp[0], sw[0], cur, actB);    // 512 -> 64
    runBlock(sbp[1], sw[1], actB, shp);    // 64 -> 32
    gn(shp, snm, shnG, shnB, 32);
    {
        int total = 3 * Bc * V;
        shape_head<<<(total + 255) / 256, 256, 0, stream>>>(
            snm, shW, shB, (float*)d_out, BV, Vp, V);
    }

    // camera path (from x512 in cur)
    gn(cur, tbuf, cnG, cnB, 512);
    {
        int total = Bc * V;
        cam_c_kernel<<<(total + 255) / 256, 256, 0, stream>>>(
            tbuf, clW, clB, cbuf, BV, Vp, V);
    }
    cam_out_kernel<<<Bc * 3, 256, 0, stream>>>(
        cbuf, fcW, fcB, (float*)d_out + (size_t)Bc * 3 * V, V);
}
